// NemotronHMambaDecoderLayer_84396107366824
// MI455X (gfx1250) — matrix operand order repaired and checked by reading, not compiled, not run on hardware
//
#include <hip/hip_runtime.h>
#include <cstdint>

// ---------------- problem constants ----------------
#define T_      2048
#define D_      4096
#define NSSM_   128
#define KC_     4
#define G_      8
#define H_      128
#define P_      64
#define INTER_  8192      // H*P
#define CONVD_  10240     // INTER + 2*G*NSSM
#define PROJ_   18560     // INTER + CONVD + H
#define QC_     128
#define NC_     16
#define EPS_    1e-5f

typedef __attribute__((ext_vector_type(16))) __bf16 v16bf;
typedef __attribute__((ext_vector_type(8)))  float  v8f;

// ---------------- WMMA helpers (gfx1250, wave32) ----------------
__device__ __forceinline__ v8f wmma_bf16(v16bf a, v16bf b, v8f c) {
    return __builtin_amdgcn_wmma_f32_16x16x32_bf16(false, a, false, b, (short)0, c, false, false);
}

// A-operand (16x32 tile), source row-major with K contiguous.
// A layout used here (contiguous k, same as B): lane m = row0+(lane&15); element e -> K = e + 16*hh
__device__ __forceinline__ v16bf frag_a(const __bf16* buf, int ld, int row0, int k0, int lane) {
    int r = row0 + (lane & 15); int hh = lane >> 4;
    const uint32_t* p32 = (const uint32_t*)(buf + r * ld + k0);
    union { v16bf v; uint32_t u[8]; } x;
#pragma unroll
    for (int e2 = 0; e2 < 8; ++e2) {
        int e = 2 * e2;
        int k = e + 16 * hh;
        x.u[e2] = p32[k >> 1];
    }
    return x.v;
}

// B-operand (32x16 tile), source rows = N index, K contiguous.
// B layout used here (contiguous k, same as A): lane n = row0+(lane&15); element e -> K = e + 16*hh
__device__ __forceinline__ v16bf frag_b(const __bf16* buf, int ld, int row0, int k0, int lane) {
    int r = row0 + (lane & 15); int hh = lane >> 4;
    const uint32_t* p32 = (const uint32_t*)(buf + r * ld + k0);
    union { v16bf v; uint32_t u[8]; } x;
#pragma unroll
    for (int e2 = 0; e2 < 8; ++e2) x.u[e2] = p32[e2 + 8 * hh];
    return x.v;
}

// B-operand from a [K x N] row-major buffer (K strided by ld)
__device__ __forceinline__ v16bf frag_bt(const __bf16* buf, int ld, int k0, int n0, int lane) {
    int n = n0 + (lane & 15); int hh = lane >> 4;
    v16bf v;
#pragma unroll
    for (int e = 0; e < 16; ++e) {
        int k = e + 16 * hh;
        v[e] = buf[(k0 + k) * ld + n];
    }
    return v;
}

// A-operand transposed: source buffer [K x M] row-major, each K row scaled.
__device__ __forceinline__ v16bf frag_at_scaled(const __bf16* buf, int ld, int k0, int m0,
                                                const float* scale, int lane) {
    int m = m0 + (lane & 15); int hh = lane >> 4;
    v16bf v;
#pragma unroll
    for (int e = 0; e < 16; ++e) {
        int kk = k0 + e + 16 * hh;
        v[e] = (__bf16)((float)buf[kk * ld + m] * scale[kk]);
    }
    return v;
}

// ---------------- small utility kernels ----------------
__global__ void k_cvt_bf16(const float* __restrict__ src, __bf16* __restrict__ dst, uint32_t n) {
    uint32_t base = blockIdx.x * 1024u + threadIdx.x;
#pragma unroll
    for (int j = 0; j < 4; ++j) {
        uint32_t i = base + j * 256u;
        if (i < n) dst[i] = (__bf16)src[i];
    }
}

// resid = hs + residual ; hsn = RMSNorm(resid)*norm_w  (bf16)
__global__ __launch_bounds__(256) void k_residnorm(const float* __restrict__ hs,
                                                   const float* __restrict__ res,
                                                   const float* __restrict__ nw,
                                                   float* __restrict__ resid_out,
                                                   __bf16* __restrict__ hsn) {
    uint32_t t = blockIdx.x, tid = threadIdx.x;
    float v[16]; float ss = 0.f;
#pragma unroll
    for (int j = 0; j < 16; ++j) {
        uint32_t i = j * 256u + tid;
        float r = hs[t * D_ + i] + res[t * D_ + i];
        v[j] = r; resid_out[t * D_ + i] = r; ss += r * r;
    }
    __shared__ float red[256];
    red[tid] = ss; __syncthreads();
    for (int s = 128; s > 0; s >>= 1) { if (tid < (uint32_t)s) red[tid] += red[tid + s]; __syncthreads(); }
    float sc = rsqrtf(red[0] / (float)D_ + EPS_);
#pragma unroll
    for (int j = 0; j < 16; ++j) {
        uint32_t i = j * 256u + tid;
        hsn[t * D_ + i] = (__bf16)(v[j] * sc * nw[i]);
    }
}

// dt softplus + cumsum per (chunk, head); also chunk decay
__global__ void k_dtcum(const float* __restrict__ proj, const float* __restrict__ dtb,
                        const float* __restrict__ Alog, float* __restrict__ dts,
                        float* __restrict__ cumb, float* __restrict__ cdec) {
    uint32_t h = threadIdx.x, c = blockIdx.x;
    float A = -expf(Alog[h]); float bias = dtb[h]; float acc = 0.f;
    for (uint32_t q = 0; q < QC_; ++q) {
        uint32_t t = c * QC_ + q;
        float raw = proj[t * PROJ_ + (INTER_ + CONVD_) + h] + bias;
        float dt = raw > 20.f ? raw : log1pf(expf(raw));
        dts[t * H_ + h] = dt;
        acc += dt * A;
        cumb[t * H_ + h] = acc;
    }
    cdec[c * H_ + h] = expf(acc);
}

// causal depthwise conv (K=4) + bias + SiLU over xBC columns of proj
__global__ __launch_bounds__(256) void k_conv(const float* __restrict__ proj,
                                              const float* __restrict__ cw,
                                              const float* __restrict__ cb,
                                              float* __restrict__ xbcs) {
    uint32_t j = blockIdx.x * 256u + threadIdx.x;
    uint32_t t = blockIdx.y;
    float acc = cb[j];
#pragma unroll
    for (int k = 0; k < KC_; ++k) {
        int tt = (int)t + k - (KC_ - 1);
        if (tt >= 0) acc += cw[j * KC_ + k] * proj[(uint32_t)tt * PROJ_ + INTER_ + j];
    }
    acc = acc / (1.f + expf(-acc));
    xbcs[t * CONVD_ + j] = acc;
}

// exclusive decayed scan over chunks for SSM state (in place)
__global__ void k_scan(float* __restrict__ Sg, const float* __restrict__ cdec) {
    uint32_t lin = blockIdx.x * 256u + threadIdx.x;   // over H*NSSM*P = 1M
    uint32_t p = lin & 63, n = (lin >> 6) & 127, h = lin >> 13;
    float s = 0.f;
#pragma unroll
    for (int c = 0; c < NC_; ++c) {
        uint32_t idx = ((c * H_ + h) * NSSM_ + n) * P_ + p;
        float tmp = Sg[idx];
        Sg[idx] = s;
        s = cdec[c * H_ + h] * s + tmp;
    }
}

// y = (Y + D*x) * silu(z); RMS over INTER; *gn_w -> bf16
__global__ __launch_bounds__(256) void k_gate(const float* __restrict__ Yg,
                                              const float* __restrict__ xbcs,
                                              const float* __restrict__ proj,
                                              const float* __restrict__ Dr,
                                              const float* __restrict__ gnw,
                                              __bf16* __restrict__ yb) {
    uint32_t t = blockIdx.x, tid = threadIdx.x;
    float v[32]; float ss = 0.f;
#pragma unroll
    for (int j = 0; j < 32; ++j) {
        uint32_t i = j * 256u + tid;
        uint32_t h = i >> 6;
        float yv = Yg[t * INTER_ + i] + Dr[h] * xbcs[t * CONVD_ + i];
        float z = proj[t * PROJ_ + i];
        float val = yv * (z / (1.f + expf(-z)));
        v[j] = val; ss += val * val;
    }
    __shared__ float red[256];
    red[tid] = ss; __syncthreads();
    for (int s = 128; s > 0; s >>= 1) { if (tid < (uint32_t)s) red[tid] += red[tid + s]; __syncthreads(); }
    float sc = rsqrtf(red[0] / (float)INTER_ + EPS_);
#pragma unroll
    for (int j = 0; j < 32; ++j) {
        uint32_t i = j * 256u + tid;
        yb[t * INTER_ + i] = (__bf16)(v[j] * sc * gnw[i]);
    }
}

// ---------------- bf16 WMMA GEMM: C[M,N] = A[M,K] * W[N,K]^T ----------------
// 128x128 block tile, 8 waves (2x4), each wave 4x2 16x16 tiles, K staged 64 at a time.
__global__ __launch_bounds__(256) void k_gemm(const __bf16* __restrict__ A,
                                              const __bf16* __restrict__ W,
                                              float* __restrict__ C,
                                              int Kd, int lda, int ldw, int ldc) {
    __shared__ __align__(16) uint32_t As[4096];   // 128 x 64 bf16
    __shared__ __align__(16) uint32_t Bs[4096];   // 128 x 64 bf16
    int tid = threadIdx.x, lane = tid & 31, w = tid >> 5;
    int wm = w >> 2, wn = w & 3;
    uint32_t m0 = blockIdx.y * 128u, n0 = blockIdx.x * 128u;
    v8f acc[4][2];
#pragma unroll
    for (int i = 0; i < 4; ++i)
#pragma unroll
        for (int j = 0; j < 2; ++j) acc[i][j] = (v8f){0,0,0,0,0,0,0,0};

    const uint32_t* Au = (const uint32_t*)A;
    const uint32_t* Wu = (const uint32_t*)W;
    uint32_t ldad = (uint32_t)lda >> 1, ldwd = (uint32_t)ldw >> 1;

    // staging: thread -> (row, half): 4x b128 global loads per matrix per stage
    uint32_t srow = (uint32_t)tid >> 1, shalf = (uint32_t)tid & 1;
    uint32_t aoff = (m0 + srow) * ldad + shalf * 16u;
    uint32_t woff = (n0 + srow) * ldwd + shalf * 16u;
    uint32_t ldst = srow * 32u + shalf * 16u;

    int nk = Kd >> 6;
    for (int kt = 0; kt < nk; ++kt) {
        const uint4* ap = (const uint4*)(Au + aoff);
        const uint4* wp = (const uint4*)(Wu + woff);
        uint4 a0 = ap[0], a1 = ap[1], a2 = ap[2], a3 = ap[3];
        uint4 b0 = wp[0], b1 = wp[1], b2 = wp[2], b3 = wp[3];
        uint4* asd = (uint4*)&As[ldst];
        uint4* bsd = (uint4*)&Bs[ldst];
        asd[0] = a0; asd[1] = a1; asd[2] = a2; asd[3] = a3;
        bsd[0] = b0; bsd[1] = b1; bsd[2] = b2; bsd[3] = b3;
        __syncthreads();
#pragma unroll
        for (int ks = 0; ks < 2; ++ks) {
            v16bf af[4];
#pragma unroll
            for (int i = 0; i < 4; ++i)
                af[i] = frag_a((const __bf16*)As, 64, wm * 64 + i * 16, ks * 32, lane);
#pragma unroll
            for (int j = 0; j < 2; ++j) {
                v16bf bf = frag_b((const __bf16*)Bs, 64, wn * 32 + j * 16, ks * 32, lane);
#pragma unroll
                for (int i = 0; i < 4; ++i) acc[i][j] = wmma_bf16(af[i], bf, acc[i][j]);
            }
        }
        __syncthreads();
        aoff += 32u; woff += 32u;
    }
    int hh = lane >> 4;
    uint32_t colb = n0 + (uint32_t)(wn * 32) + (uint32_t)(lane & 15);
    uint32_t rowb = m0 + (uint32_t)(wm * 64) + (uint32_t)(8 * hh);
#pragma unroll
    for (int i = 0; i < 4; ++i)
#pragma unroll
        for (int j = 0; j < 2; ++j)
#pragma unroll
            for (int r = 0; r < 8; ++r) {
                uint32_t row = rowb + (uint32_t)(i * 16 + r);
                uint32_t col = colb + (uint32_t)(j * 16);
                C[row * (uint32_t)ldc + col] = acc[i][j][r];
            }
}

// ---------------- SSM kernel 1: per (chunk, head) intra-chunk + local state ----------------
// grid: (H, NC), 256 threads. Dynamic LDS:
//   Cs[128*128] bf16 (C, later CBL), Bs[128*128] bf16, Xs[128*64] bf16, cum[128] f32, dec[128] f32
__global__ __launch_bounds__(256) void k_ssm1(const float* __restrict__ xbcs,
                                              const float* __restrict__ dts,
                                              const float* __restrict__ cumb,
                                              float* __restrict__ Yg,
                                              float* __restrict__ Sg) {
    extern __shared__ char smem[];
    __bf16* Cs = (__bf16*)smem;
    __bf16* Bsm = Cs + 128 * 128;
    __bf16* Xs = Bsm + 128 * 128;
    float* cum_s = (float*)(Xs + 128 * 64);
    float* dec_s = cum_s + 128;

    uint32_t h = blockIdx.x, c = blockIdx.y;
    uint32_t g = h >> 4;                       // H/G = 16 heads per group
    int tid = threadIdx.x, lane = tid & 31, w = tid >> 5, hh = lane >> 4;

    // ---- stage operands (fp32 -> bf16) ----
#pragma unroll
    for (int i = 0; i < 32; ++i) {             // Xd = dt * x  (128 x 64)
        uint32_t lin = tid + 256u * i;
        uint32_t q = lin >> 6, p = lin & 63;
        uint32_t t = c * QC_ + q;
        Xs[lin] = (__bf16)(xbcs[t * CONVD_ + h * P_ + p] * dts[t * H_ + h]);
    }
#pragma unroll
    for (int i = 0; i < 64; ++i) {             // B, C (128 x 128)
        uint32_t lin = tid + 256u * i;
        uint32_t q = lin >> 7, n = lin & 127;
        uint32_t t = c * QC_ + q;
        Bsm[lin] = (__bf16)(xbcs[t * CONVD_ + INTER_ + g * NSSM_ + n]);
        Cs[lin]  = (__bf16)(xbcs[t * CONVD_ + INTER_ + G_ * NSSM_ + g * NSSM_ + n]);
    }
    if (tid < 128) cum_s[tid] = cumb[(c * QC_ + tid) * H_ + h];
    __syncthreads();
    if (tid < 128) dec_s[tid] = expf(cum_s[127] - cum_s[tid]);
    __syncthreads();

    int q0 = w * 16;

    // ---- phase B: CB = C * B^T  (128x128), then mask/decay -> CBL into Cs ----
    v16bf af[4];
#pragma unroll
    for (int ks = 0; ks < 4; ++ks) af[ks] = frag_a(Cs, 128, q0, ks * 32, lane);
    v8f cb[8];
#pragma unroll
    for (int jt = 0; jt < 8; ++jt) cb[jt] = (v8f){0,0,0,0,0,0,0,0};
#pragma unroll
    for (int jt = 0; jt < 8; ++jt)
#pragma unroll
        for (int ks = 0; ks < 4; ++ks) {
            v16bf bf = frag_b(Bsm, 128, jt * 16, ks * 32, lane);
            cb[jt] = wmma_bf16(af[ks], bf, cb[jt]);
        }
    __syncthreads();                           // everyone done reading C
#pragma unroll
    for (int jt = 0; jt < 8; ++jt)
#pragma unroll
        for (int r = 0; r < 8; ++r) {
            int q = q0 + r + 8 * hh;
            int k = jt * 16 + (lane & 15);
            float vv = cb[jt][r];
            vv = (q >= k) ? vv * expf(cum_s[q] - cum_s[k]) : 0.f;
            Cs[q * 128 + k] = (__bf16)vv;      // CBL replaces C
        }
    __syncthreads();

    // ---- phase C: Y1 = CBL * Xd  (128 x 64) ----
#pragma unroll
    for (int ks = 0; ks < 4; ++ks) af[ks] = frag_a(Cs, 128, q0, ks * 32, lane);
    v8f yacc[4];
#pragma unroll
    for (int pt = 0; pt < 4; ++pt) yacc[pt] = (v8f){0,0,0,0,0,0,0,0};
#pragma unroll
    for (int pt = 0; pt < 4; ++pt)
#pragma unroll
        for (int ks = 0; ks < 4; ++ks) {
            v16bf bf = frag_bt(Xs, 64, ks * 32, pt * 16, lane);
            yacc[pt] = wmma_bf16(af[ks], bf, yacc[pt]);
        }
#pragma unroll
    for (int pt = 0; pt < 4; ++pt)
#pragma unroll
        for (int r = 0; r < 8; ++r) {
            uint32_t t = c * QC_ + (uint32_t)(q0 + r + 8 * hh);
            uint32_t p = (uint32_t)(pt * 16 + (lane & 15));
            Yg[t * INTER_ + h * P_ + p] = yacc[pt][r];
        }

    // ---- phase D: S_loc = (B*dec)^T * Xd  (128 x 64) ----
#pragma unroll
    for (int ks = 0; ks < 4; ++ks) af[ks] = frag_at_scaled(Bsm, 128, ks * 32, w * 16, dec_s, lane);
    v8f sacc[4];
#pragma unroll
    for (int pt = 0; pt < 4; ++pt) sacc[pt] = (v8f){0,0,0,0,0,0,0,0};
#pragma unroll
    for (int pt = 0; pt < 4; ++pt)
#pragma unroll
        for (int ks = 0; ks < 4; ++ks) {
            v16bf bf = frag_bt(Xs, 64, ks * 32, pt * 16, lane);
            sacc[pt] = wmma_bf16(af[ks], bf, sacc[pt]);
        }
#pragma unroll
    for (int pt = 0; pt < 4; ++pt)
#pragma unroll
        for (int r = 0; r < 8; ++r) {
            uint32_t n = (uint32_t)(w * 16 + r + 8 * hh);
            uint32_t p = (uint32_t)(pt * 16 + (lane & 15));
            Sg[((c * H_ + h) * NSSM_ + n) * P_ + p] = sacc[pt][r];
        }
}

// ---------------- SSM kernel 2: Y += (C * exp(cum)) * S_prev ----------------
// Dynamic LDS: Ce[128*128] bf16, Ss[128*64] bf16, cum[128] f32
__global__ __launch_bounds__(256) void k_ssm2(const float* __restrict__ xbcs,
                                              const float* __restrict__ cumb,
                                              const float* __restrict__ Sg,
                                              float* __restrict__ Yg) {
    extern __shared__ char smem[];
    __bf16* Ce = (__bf16*)smem;
    __bf16* Ss = Ce + 128 * 128;
    float* cum_s = (float*)(Ss + 128 * 64);

    uint32_t h = blockIdx.x, c = blockIdx.y;
    uint32_t g = h >> 4;
    int tid = threadIdx.x, lane = tid & 31, w = tid >> 5, hh = lane >> 4;

    if (tid < 128) cum_s[tid] = cumb[(c * QC_ + tid) * H_ + h];
    __syncthreads();
#pragma unroll
    for (int i = 0; i < 64; ++i) {
        uint32_t lin = tid + 256u * i;
        uint32_t q = lin >> 7, n = lin & 127;
        uint32_t t = c * QC_ + q;
        Ce[lin] = (__bf16)(xbcs[t * CONVD_ + INTER_ + G_ * NSSM_ + g * NSSM_ + n] *
                           expf(cum_s[q]));
    }
#pragma unroll
    for (int i = 0; i < 32; ++i) {
        uint32_t lin = tid + 256u * i;
        uint32_t n = lin >> 6, p = lin & 63;
        Ss[lin] = (__bf16)(Sg[((c * H_ + h) * NSSM_ + n) * P_ + p]);
    }
    __syncthreads();

    int q0 = w * 16;
    v16bf af[4];
#pragma unroll
    for (int ks = 0; ks < 4; ++ks) af[ks] = frag_a(Ce, 128, q0, ks * 32, lane);
    v8f yacc[4];
#pragma unroll
    for (int pt = 0; pt < 4; ++pt) yacc[pt] = (v8f){0,0,0,0,0,0,0,0};
#pragma unroll
    for (int pt = 0; pt < 4; ++pt)
#pragma unroll
        for (int ks = 0; ks < 4; ++ks) {
            v16bf bf = frag_bt(Ss, 64, ks * 32, pt * 16, lane);
            yacc[pt] = wmma_bf16(af[ks], bf, yacc[pt]);
        }
#pragma unroll
    for (int pt = 0; pt < 4; ++pt)
#pragma unroll
        for (int r = 0; r < 8; ++r) {
            uint32_t t = c * QC_ + (uint32_t)(q0 + r + 8 * hh);
            uint32_t p = (uint32_t)(pt * 16 + (lane & 15));
            uint32_t idx = t * INTER_ + h * P_ + p;
            Yg[idx] = Yg[idx] + yacc[pt][r];
        }
}

// ---------------- host: workspace layout + launches ----------------
static inline size_t align256(size_t x) { return (x + 255) & ~(size_t)255; }

extern "C" void kernel_launch(void* const* d_in, const int* in_sizes, int n_in,
                              void* d_out, int out_size, void* d_ws, size_t ws_size,
                              hipStream_t stream) {
    (void)in_sizes; (void)n_in; (void)out_size; (void)ws_size;
    const float* hs      = (const float*)d_in[0];
    const float* resi    = (const float*)d_in[1];
    const float* norm_w  = (const float*)d_in[2];
    const float* in_w    = (const float*)d_in[3];
    const float* conv_w  = (const float*)d_in[4];
    const float* conv_b  = (const float*)d_in[5];
    const float* A_log   = (const float*)d_in[6];
    const float* D_res   = (const float*)d_in[7];
    const float* dt_bias = (const float*)d_in[8];
    const float* gn_w    = (const float*)d_in[9];
    const float* out_w   = (const float*)d_in[10];

    float* out  = (float*)d_out;                 // T x D
    float* resid_out = out + (size_t)T_ * D_;    // T x D

    char* ws = (char*)d_ws;
    size_t o = 0;
    __bf16* hsn   = (__bf16*)(ws + o); o = align256(o + (size_t)T_ * D_ * 2);
    __bf16* w_in  = (__bf16*)(ws + o); o = align256(o + (size_t)PROJ_ * D_ * 2);
    __bf16* w_out = (__bf16*)(ws + o); o = align256(o + (size_t)D_ * INTER_ * 2);
    float*  proj  = (float*) (ws + o); o = align256(o + (size_t)T_ * PROJ_ * 4);
    float*  xbcs  = (float*) (ws + o); o = align256(o + (size_t)T_ * CONVD_ * 4);
    float*  dts   = (float*) (ws + o); o = align256(o + (size_t)T_ * H_ * 4);
    float*  cumb  = (float*) (ws + o); o = align256(o + (size_t)T_ * H_ * 4);
    float*  cdec  = (float*) (ws + o); o = align256(o + (size_t)NC_ * H_ * 4);
    float*  Sg    = (float*) (ws + o); o = align256(o + (size_t)NC_ * H_ * NSSM_ * P_ * 4);
    float*  Yg    = (float*) (ws + o); o = align256(o + (size_t)T_ * INTER_ * 4);
    __bf16* yb    = (__bf16*)(ws + o); o = align256(o + (size_t)T_ * INTER_ * 2);

    // 1) weights -> bf16 (in_proj bf16 = 152 MB, fits MI455X 192 MB L2)
    {
        uint32_t n1 = (uint32_t)((size_t)PROJ_ * D_);
        k_cvt_bf16<<<(n1 + 1023) / 1024, 256, 0, stream>>>(in_w, w_in, n1);
        uint32_t n2 = (uint32_t)((size_t)D_ * INTER_);
        k_cvt_bf16<<<(n2 + 1023) / 1024, 256, 0, stream>>>(out_w, w_out, n2);
    }
    // 2) residual add + RMSNorm (emits resid output)
    k_residnorm<<<T_, 256, 0, stream>>>(hs, resi, norm_w, resid_out, hsn);
    // 3) in_proj GEMM: [2048 x 18560] = hsn[2048x4096] * w_in[18560x4096]^T
    k_gemm<<<dim3(PROJ_ / 128, T_ / 128), 256, 0, stream>>>(hsn, w_in, proj, D_, D_, D_, PROJ_);
    // 4) dt softplus + per-chunk cumsum + chunk decay
    k_dtcum<<<NC_, H_, 0, stream>>>(proj, dt_bias, A_log, dts, cumb, cdec);
    // 5) causal conv + SiLU
    k_conv<<<dim3(CONVD_ / 256, T_), 256, 0, stream>>>(proj, conv_w, conv_b, xbcs);
    // 6) SSM intra-chunk: Y1 and S_loc
    {
        size_t sh = (size_t)(128 * 128 + 128 * 128 + 128 * 64) * 2 + 256 * 4;  // 82944 B
        k_ssm1<<<dim3(H_, NC_), 256, sh, stream>>>(xbcs, dts, cumb, Yg, Sg);
    }
    // 7) inter-chunk state scan (exclusive, decayed) in place
    k_scan<<<(H_ * NSSM_ * P_) / 256, 256, 0, stream>>>(Sg, cdec);
    // 8) SSM inter-chunk contribution: Y += Ce * S_prev
    {
        size_t sh = (size_t)(128 * 128 + 128 * 64) * 2 + 128 * 4;              // 49664 B
        k_ssm2<<<dim3(H_, NC_), 256, sh, stream>>>(xbcs, cumb, Sg, Yg);
    }
    // 9) gating + D skip + RMS + gn_w -> bf16
    k_gate<<<T_, 256, 0, stream>>>(Yg, xbcs, proj, D_res, gn_w, yb);
    // 10) out_proj GEMM: out[2048x4096] = yb[2048x8192] * w_out[4096x8192]^T
    k_gemm<<<dim3(D_ / 128, T_ / 128), 256, 0, stream>>>(yb, w_out, out, INTER_, INTER_, INTER_, D_);
}
